// Layer_2310692405778
// MI455X (gfx1250) — matrix operand order repaired and checked by reading, not compiled, not run on hardware
//
#include <hip/hip_runtime.h>

// ---------------------------------------------------------------------------
// leaky_relu(bias + x @ densify(W))  on MI455X (gfx1250, wave32, WMMA)
//
// bf16x3 WMMA GEMM (v_wmma_f32_16x16x32_bf16, fp32 accumulate):
//   x@W ~= xh@wh + xl@wh + xh@wl        (near-fp32 accuracy)
// W densified TRANSPOSED (Wt[n][k]) so all fragments are contiguous
// global_load_b128s in the wave32 WMMA operand layout.
//
// Round-3 change: per-wave tile 32x64 -> 64x64 (4 A tiles x 4 B tiles,
// 16 fp32 accumulators).  Loads per k-step: 4*(4+4)=32 b128 for 48 WMMAs
// (0.67 loads/WMMA, 341 B/WMMA, was 1.0 / 512) — the L2-resident GEMM is
// L0->VGPR bandwidth bound, so bytes/WMMA is the figure of merit.
// Block = 4 waves in 2x2 covering 128x128 for WGP$-level operand sharing.
// ---------------------------------------------------------------------------

#define IN_D   4096
#define OUT_D  4096
#define BATCH  4096
#define NNZ_N  1677722
#define NEG_SLOPE 0.1f

typedef __bf16 v16bf __attribute__((ext_vector_type(16)));
typedef __bf16 v8bf  __attribute__((ext_vector_type(8)));
typedef float  v8f   __attribute__((ext_vector_type(8)));

union V16 { v16bf v; v8bf h[2]; };

// ---------------------------------------------------------------------------
// 1) Zero the dense W (hi+lo) region: 64 MB = 4Mi uint4
// ---------------------------------------------------------------------------
__global__ void zero_ws_kernel(uint4* __restrict__ p, long long nquads) {
    long long i = (long long)blockIdx.x * blockDim.x + threadIdx.x;
    if (i < nquads) {
        uint4 z; z.x = 0u; z.y = 0u; z.z = 0u; z.w = 0u;
        p[i] = z;
    }
}

// ---------------------------------------------------------------------------
// 2) Scatter COO weights into TRANSPOSED dense W:  Wt[n][k] = w  (hi/lo bf16)
// ---------------------------------------------------------------------------
__global__ void scatter_w_kernel(const float* __restrict__ weight,
                                 const int*   __restrict__ rows,
                                 const int*   __restrict__ cols,
                                 unsigned short* __restrict__ whi_,
                                 unsigned short* __restrict__ wlo_,
                                 int nnz) {
    int i = blockIdx.x * blockDim.x + threadIdx.x;
    if (i >= nnz) return;
    __bf16* whi = reinterpret_cast<__bf16*>(whi_);
    __bf16* wlo = reinterpret_cast<__bf16*>(wlo_);
    int r = rows[i];          // k index (input dim)
    int c = cols[i];          // n index (output dim)
    float f = weight[i];
    __bf16 h = (__bf16)f;
    __bf16 l = (__bf16)(f - (float)h);
    size_t idx = (size_t)c * IN_D + (size_t)r;   // transposed: n-major, k-minor
    whi[idx] = h;
    wlo[idx] = l;
}

// ---------------------------------------------------------------------------
// 3) Split x (fp32, row-major M x K) into bf16 hi/lo, same layout
// ---------------------------------------------------------------------------
__global__ void split_x_kernel(const float* __restrict__ src,
                               unsigned short* __restrict__ hi_,
                               unsigned short* __restrict__ lo_,
                               int n) {
    int i = blockIdx.x * blockDim.x + threadIdx.x;
    if (i >= n) return;
    __bf16* hi = reinterpret_cast<__bf16*>(hi_);
    __bf16* lo = reinterpret_cast<__bf16*>(lo_);
    float f = src[i];
    __bf16 h = (__bf16)f;
    hi[i] = h;
    lo[i] = (__bf16)(f - (float)h);
}

// ---------------------------------------------------------------------------
// 4) WMMA GEMM.  One wave owns a 64(M) x 64(N) tile: 4 A fragments x 4 B
//    fragments = 16 fp32 C fragments.  Fragment addressing (wave32):
//    A (16x32 bf16): lane m = lane&15; per lane two contiguous 8-bf16 chunks
//      at k = kb + (lane>>4)*8 and +16 in row-major x.
//    B (32x16 bf16): lane n = lane&15; per lane two contiguous 8-bf16 chunks
//      at k = kb + (lane>>4)*8 and +16 in transposed Wt[n][k].
// ---------------------------------------------------------------------------
__global__ __launch_bounds__(128)
void gemm_bf16x3_kernel(const unsigned short* __restrict__ xhi_,
                        const unsigned short* __restrict__ xlo_,
                        const unsigned short* __restrict__ whi_,
                        const unsigned short* __restrict__ wlo_,
                        const float* __restrict__ bias,
                        float* __restrict__ out) {
    const __bf16* Xhi = reinterpret_cast<const __bf16*>(xhi_);
    const __bf16* Xlo = reinterpret_cast<const __bf16*>(xlo_);
    const __bf16* Whi = reinterpret_cast<const __bf16*>(whi_);
    const __bf16* Wlo = reinterpret_cast<const __bf16*>(wlo_);

    const int lane = threadIdx.x & 31;
    const int wave = threadIdx.x >> 5;          // 0..3

    // Block covers 128(M) x 128(N); waves arranged 2x2 within it.
    const int NB_N = OUT_D / 128;               // 32
    const int bm = blockIdx.x / NB_N;           // 0..31
    const int bn = blockIdx.x % NB_N;           // 0..31
    const int m_base = bm * 128 + (wave >> 1) * 64;
    const int n_base = bn * 128 + (wave & 1) * 64;

    const int hl  = lane >> 4;                  // half-wave select
    const int l16 = lane & 15;

    // Per-lane base pointers.  Row/column tiles are 16*IN_D*2 = 128KB apart;
    // the compiler folds those into b128 immediate offsets where possible.
    const __bf16* xh_row = Xhi + (size_t)(m_base + l16) * IN_D;
    const __bf16* xl_row = Xlo + (size_t)(m_base + l16) * IN_D;
    const __bf16* wh_col = Whi + (size_t)(n_base + l16) * IN_D;
    const __bf16* wl_col = Wlo + (size_t)(n_base + l16) * IN_D;
    const size_t TSTRIDE = (size_t)16 * IN_D;   // 16 rows/cols in elements

    v8f acc[16] = {};                           // [mt*4 + t]

#pragma unroll 1
    for (int kb = 0; kb < IN_D; kb += 32) {
        const int ka0 = kb + hl * 8;            // A chunk 0 (frag VGPRs 0-3)
        const int ka1 = ka0 + 16;               // A chunk 1 (frag VGPRs 4-7)
        V16 ah[4], al[4];
#pragma unroll
        for (int mt = 0; mt < 4; ++mt) {
            const __bf16* xh = xh_row + (size_t)mt * TSTRIDE;
            const __bf16* xl = xl_row + (size_t)mt * TSTRIDE;
            ah[mt].h[0] = *(const v8bf*)(xh + ka0);
            ah[mt].h[1] = *(const v8bf*)(xh + ka1);
            al[mt].h[0] = *(const v8bf*)(xl + ka0);
            al[mt].h[1] = *(const v8bf*)(xl + ka1);
        }

        const int kb0 = kb + hl * 8;            // B: two 8-K chunks, same as A
#pragma unroll
        for (int t = 0; t < 4; ++t) {
            const __bf16* wh = wh_col + (size_t)t * TSTRIDE;
            const __bf16* wl = wl_col + (size_t)t * TSTRIDE;
            V16 bh, bl;
            bh.h[0] = *(const v8bf*)(wh + kb0);
            bh.h[1] = *(const v8bf*)(wh + kb0 + 16);
            bl.h[0] = *(const v8bf*)(wl + kb0);
            bl.h[1] = *(const v8bf*)(wl + kb0 + 16);
#pragma unroll
            for (int mt = 0; mt < 4; ++mt) {
                // bf16x3: xh*wh + xl*wh + xh*wl, fp32 accumulate
                acc[mt * 4 + t] = __builtin_amdgcn_wmma_f32_16x16x32_bf16(
                    false, ah[mt].v, false, bh.v, (short)0, acc[mt * 4 + t],
                    false, false);
                acc[mt * 4 + t] = __builtin_amdgcn_wmma_f32_16x16x32_bf16(
                    false, al[mt].v, false, bh.v, (short)0, acc[mt * 4 + t],
                    false, false);
                acc[mt * 4 + t] = __builtin_amdgcn_wmma_f32_16x16x32_bf16(
                    false, ah[mt].v, false, bl.v, (short)0, acc[mt * 4 + t],
                    false, false);
            }
        }
    }

    // Epilogue: C/D layout — lane half selects M block (rows +0..7 / +8..15),
    // VGPR r is the row within the block, N = lane&15 within the tile.
#pragma unroll
    for (int t = 0; t < 4; ++t) {
        const int n = n_base + t * 16 + l16;
        const float b = bias[n];
#pragma unroll
        for (int mt = 0; mt < 4; ++mt) {
            const int mrow = m_base + mt * 16 + hl * 8;
#pragma unroll
            for (int r = 0; r < 8; ++r) {
                float v = acc[mt * 4 + t][r] + b;
                v = (v > 0.0f) ? v : NEG_SLOPE * v;
                out[(size_t)(mrow + r) * OUT_D + n] = v;
            }
        }
    }
}

// ---------------------------------------------------------------------------
// Launcher.  Workspace layout (128 MB):
//   [0,32M)    Wt_hi  bf16  (transposed dense W, n-major)
//   [32M,64M)  Wt_lo  bf16
//   [64M,96M)  X_hi   bf16  (row-major M x K)
//   [96M,128M) X_lo   bf16
// ---------------------------------------------------------------------------
extern "C" void kernel_launch(void* const* d_in, const int* in_sizes, int n_in,
                              void* d_out, int out_size, void* d_ws, size_t ws_size,
                              hipStream_t stream) {
    const float* x      = (const float*)d_in[0];
    const float* weight = (const float*)d_in[1];
    const float* bias   = (const float*)d_in[2];
    const int*   rows   = (const int*)d_in[3];
    const int*   cols   = (const int*)d_in[4];
    float*       out    = (float*)d_out;

    char* ws = (char*)d_ws;
    const size_t MB32 = (size_t)32 << 20;
    unsigned short* Whi = (unsigned short*)(ws);
    unsigned short* Wlo = (unsigned short*)(ws + MB32);
    unsigned short* Xhi = (unsigned short*)(ws + 2 * MB32);
    unsigned short* Xlo = (unsigned short*)(ws + 3 * MB32);

    // 1) zero W hi+lo (64 MB = 4Mi uint4)
    long long nquads = (long long)(2 * MB32) / 16;
    zero_ws_kernel<<<(int)(nquads / 256), 256, 0, stream>>>((uint4*)ws, nquads);

    // 2) split x into bf16 hi/lo
    int nx = BATCH * IN_D;
    split_x_kernel<<<(nx + 255) / 256, 256, 0, stream>>>(x, Xhi, Xlo, nx);

    // 3) scatter weights into transposed dense W (after zeroing, same stream)
    scatter_w_kernel<<<(NNZ_N + 255) / 256, 256, 0, stream>>>(
        weight, rows, cols, Whi, Wlo, NNZ_N);

    // 4) WMMA GEMM + bias + leaky relu
    //    grid: (BATCH/128) * (OUT_D/128) = 32 * 32 = 1024 blocks of 4 waves
    int blocks = (BATCH / 128) * (OUT_D / 128);
    gemm_bf16x3_kernel<<<blocks, 128, 0, stream>>>(Xhi, Xlo, Whi, Wlo, bias, out);
}
